// DANetHead_66803921322720
// MI455X (gfx1250) — matrix operand order repaired and checked by reading, not compiled, not run on hardware
//
#include <hip/hip_runtime.h>
#include <hip/hip_bf16.h>

typedef __attribute__((ext_vector_type(16))) __bf16 v16bf;
typedef __attribute__((ext_vector_type(8)))  __bf16 v8bf;
typedef __attribute__((ext_vector_type(8)))  float  v8f;

#define NPIX   4096
#define CI_    512
#define C8_    64
#define NC_    19
#define CIN_   2048
#define BATCH_ 2

// LDS strides (elements): chosen so rows are 16B-aligned for b128 access.
#define SA_MK  40    // A as [m][k]   (64 x 32 + pad)
#define SA_KM  72    // A as [k][m]   (32 x 64 + pad)
#define SB_NK  40    // B as [n][k]   (128 x 32 + pad)
#define SB_KN  136   // B as [k][n]   (32 x 128 + pad)
#define ASZ    2560  // max(64*40, 32*72)
#define BSZ    5120  // max(128*40, 32*136)

// ---------------------------------------------------------------------------
// WMMA + fragment helpers (wave32, 16x16x32 bf16, f32 accumulate)
// ---------------------------------------------------------------------------
__device__ __forceinline__ v8f wmma_bf16(v16bf a, v16bf b, v8f c) {
  return __builtin_amdgcn_wmma_f32_16x16x32_bf16(false, a, false, b, (short)0, c,
                                                 false, false);
}

// A fragment from LDS [m][k] (stride SA_MK). Lane L: row M = base+(L&15);
// contiguous k: elems 0-15 -> k = h16..h16+15
// Both halves are contiguous 16B runs -> ds_load_b128.
__device__ __forceinline__ v16bf frag_a_mk(const __bf16* s, int rowBase, int lane) {
  const __bf16* row = s + (rowBase + (lane & 15)) * SA_MK;
  const int h16 = (lane >> 4) * 16;
  v8bf lo = *(const v8bf*)(row + h16);
  v8bf hi = *(const v8bf*)(row + h16 + 8);
  return __builtin_shufflevector(lo, hi, 0, 1, 2, 3, 4, 5, 6, 7,
                                 8, 9, 10, 11, 12, 13, 14, 15);
}

// A fragment from LDS [k][m] (stride SA_KM) -- transA staging layout.
__device__ __forceinline__ v16bf frag_a_km(const __bf16* s, int mBase, int lane) {
  const int m = mBase + (lane & 15);
  const int h16 = (lane >> 4) * 16;
  v16bf a;
#pragma unroll
  for (int v = 0; v < 8; ++v) {
    int k = h16 + 2 * v;
    a[2 * v]     = s[k * SA_KM + m];
    a[2 * v + 1] = s[(k + 1) * SA_KM + m];
  }
  return a;
}

// B fragment from LDS [n][k] (stride SB_NK). Lane L: col N = base+(L&15);
// elems = k h16..h16+15 -> one contiguous 32B run -> 2x ds_load_b128.
__device__ __forceinline__ v16bf frag_b_nk(const __bf16* s, int nBase, int lane) {
  const __bf16* row = s + (nBase + (lane & 15)) * SB_NK;
  const int h16 = (lane >> 4) * 16;
  v8bf lo = *(const v8bf*)(row + h16);
  v8bf hi = *(const v8bf*)(row + h16 + 8);
  return __builtin_shufflevector(lo, hi, 0, 1, 2, 3, 4, 5, 6, 7,
                                 8, 9, 10, 11, 12, 13, 14, 15);
}

// B fragment from LDS [k][n] (stride SB_KN) -- natural for [k][n] global B.
__device__ __forceinline__ v16bf frag_b_kn(const __bf16* s, int nBase, int lane) {
  const int n = nBase + (lane & 15);
  const int h16 = (lane >> 4) * 16;
  v16bf b;
#pragma unroll
  for (int v = 0; v < 8; ++v) {
    int k = h16 + 2 * v;
    b[2 * v]     = s[k * SB_KN + n];
    b[2 * v + 1] = s[(k + 1) * SB_KN + n];
  }
  return b;
}

// C/D: VGPR r -> M = r + 8*(lane/16); N = lane&15.
__device__ __forceinline__ void store_acc(const v8f& acc, float* Cf, __bf16* Cb,
                                          const float* bias, int Mstore, int ldc,
                                          int mBase, int nBase, int lane) {
  const int half = lane >> 4, col = lane & 15;
  const int nOut = nBase + col;
#pragma unroll
  for (int r = 0; r < 8; ++r) {
    int mOut = mBase + half * 8 + r;
    if (mOut < Mstore) {
      float v = acc[r];
      if (bias) v += bias[mOut];
      if (Cf) Cf[(long long)mOut * ldc + nOut] = v;
      if (Cb) Cb[(long long)mOut * ldc + nOut] = (__bf16)v;
    }
  }
}

// ---------------------------------------------------------------------------
// Tiled bf16 GEMM, guard-free staging (all dims padded to tile multiples),
// double-buffered LDS, 64x128 block tile, 8 waves x (32x32 tile = 4 WMMA).
// TA: A global is [k][m] (fetch transposed). TB: B global is [n][k].
// ---------------------------------------------------------------------------
template <bool TA, bool TB>
__global__ __launch_bounds__(256) void k_gemm_bf16(
    const __bf16* __restrict__ A, const __bf16* __restrict__ B,
    float* __restrict__ Cf, __bf16* __restrict__ Cb,
    const float* __restrict__ bias,
    int Mstore, int K, int lda, int ldb, int ldc,
    long long sA, long long sB, long long sC) {
  __shared__ __bf16 As[2][ASZ];
  __shared__ __bf16 Bs[2][BSZ];
  const int bz = blockIdx.z;
  A += (long long)bz * sA;
  B += (long long)bz * sB;
  if (Cf) Cf += (long long)bz * sC;
  if (Cb) Cb += (long long)bz * sC;
  const int mBlock = blockIdx.y * 64;
  const int nBlock = blockIdx.x * 128;
  const int tid = threadIdx.x, lane = tid & 31, wave = tid >> 5;
  const int wm = (wave >> 2) * 32, wn = (wave & 3) * 32;

  auto stageA = [&](int buf, int k0) {
    if constexpr (!TA) {
      int m = tid >> 2, k8 = (tid & 3) * 8;                       // 64 x 4 slots
      v8bf v = *(const v8bf*)(A + (long long)(mBlock + m) * lda + k0 + k8);
      *(v8bf*)(As[buf] + m * SA_MK + k8) = v;
    } else {
      int k = tid >> 3, m8 = (tid & 7) * 8;                       // 32 x 8 slots
      v8bf v = *(const v8bf*)(A + (long long)(k0 + k) * lda + mBlock + m8);
      *(v8bf*)(As[buf] + k * SA_KM + m8) = v;
    }
  };
  auto stageB = [&](int buf, int k0) {
#pragma unroll
    for (int j = 0; j < 2; ++j) {
      int idx = tid + j * 256;                                    // 512 slots
      if constexpr (TB) {
        int n = idx >> 2, k8 = (idx & 3) * 8;
        v8bf v = *(const v8bf*)(B + (long long)(nBlock + n) * ldb + k0 + k8);
        *(v8bf*)(Bs[buf] + n * SB_NK + k8) = v;
      } else {
        int k = idx >> 4, n8 = (idx & 15) * 8;
        v8bf v = *(const v8bf*)(B + (long long)(k0 + k) * ldb + nBlock + n8);
        *(v8bf*)(Bs[buf] + k * SB_KN + n8) = v;
      }
    }
  };

  v8f acc00 = {}, acc01 = {}, acc10 = {}, acc11 = {};
  stageA(0, 0);
  stageB(0, 0);
  int buf = 0;
  for (int k0 = 0; k0 < K; k0 += 32) {
    __syncthreads();
    if (k0 + 32 < K) { stageA(buf ^ 1, k0 + 32); stageB(buf ^ 1, k0 + 32); }
    v16bf a0, a1, b0, b1;
    if constexpr (TA) {
      a0 = frag_a_km(As[buf], wm, lane);
      a1 = frag_a_km(As[buf], wm + 16, lane);
    } else {
      a0 = frag_a_mk(As[buf], wm, lane);
      a1 = frag_a_mk(As[buf], wm + 16, lane);
    }
    if constexpr (TB) {
      b0 = frag_b_nk(Bs[buf], wn, lane);
      b1 = frag_b_nk(Bs[buf], wn + 16, lane);
    } else {
      b0 = frag_b_kn(Bs[buf], wn, lane);
      b1 = frag_b_kn(Bs[buf], wn + 16, lane);
    }
    acc00 = wmma_bf16(a0, b0, acc00);
    acc01 = wmma_bf16(a0, b1, acc01);
    acc10 = wmma_bf16(a1, b0, acc10);
    acc11 = wmma_bf16(a1, b1, acc11);
    buf ^= 1;
  }
  store_acc(acc00, Cf, Cb, bias, Mstore, ldc, mBlock + wm,      nBlock + wn,      lane);
  store_acc(acc01, Cf, Cb, bias, Mstore, ldc, mBlock + wm,      nBlock + wn + 16, lane);
  store_acc(acc10, Cf, Cb, bias, Mstore, ldc, mBlock + wm + 16, nBlock + wn,      lane);
  store_acc(acc11, Cf, Cb, bias, Mstore, ldc, mBlock + wm + 16, nBlock + wn + 16, lane);
}

// ---------------------------------------------------------------------------
// Implicit-GEMM 3x3 conv (pad 1) + conv-bias + BN + ReLU, dual f32/bf16 out.
// Weights pre-transposed to bf16 [tap][outC(=512)][inC]. Double-buffered.
// ---------------------------------------------------------------------------
__global__ __launch_bounds__(256) void k_conv3x3_bnrelu(
    const float* __restrict__ X, const __bf16* __restrict__ Wt,
    const float* __restrict__ bias,
    const float* __restrict__ bn_g, const float* __restrict__ bn_b,
    const float* __restrict__ bn_m, const float* __restrict__ bn_v,
    float* __restrict__ outF, __bf16* __restrict__ outB, int inC, int nsSh) {
  __shared__ __bf16 As[2][ASZ];
  __shared__ __bf16 Bs[2][BSZ];
  const int bz = blockIdx.z;
  const float* Xb = X + (long long)bz * inC * NPIX;
  const int mBlock = blockIdx.y * 64;
  const int nBlock = blockIdx.x * 128;
  const int tid = threadIdx.x, lane = tid & 31, wave = tid >> 5;
  const int wm = (wave >> 2) * 32, wn = (wave & 3) * 32;
  const int nsteps = 1 << nsSh;
  const int total = 9 * nsteps;

  auto stage = [&](int buf, int t) {
    const int tap = t >> nsSh;
    const int kc = (t & (nsteps - 1)) << 5;
    const int dy = tap / 3 - 1, dx = tap % 3 - 1;
    // A: weights [tap][oc][ic], k-contiguous
    {
      int m = tid >> 2, k8 = (tid & 3) * 8;
      v8bf v = *(const v8bf*)(Wt + ((long long)tap * CI_ + mBlock + m) * inC + kc + k8);
      *(v8bf*)(As[buf] + m * SA_MK + k8) = v;
    }
    // B: shifted input pixels, [k][n] layout (n-contiguous in memory)
#pragma unroll
    for (int j = 0; j < 2; ++j) {
      int idx = tid + j * 256;
      int k = idx >> 4, n8 = (idx & 15) * 8;
      int ic = kc + k;
      int p = nBlock + n8;
      int h = (p >> 6) + dy;
      int w0 = (p & 63) + dx;
      const bool rowok = (unsigned)h < 64u;
      float t8[8];
      if (rowok && w0 >= 0 && w0 + 7 < 64) {
        const float* src = Xb + ((long long)ic << 12) + (h << 6) + w0;
#pragma unroll
        for (int e = 0; e < 8; ++e) t8[e] = src[e];
      } else {
#pragma unroll
        for (int e = 0; e < 8; ++e) {
          int wj = w0 + e;
          t8[e] = (rowok && (unsigned)wj < 64u)
                      ? Xb[((long long)ic << 12) + (h << 6) + wj] : 0.0f;
        }
      }
      v8bf vv;
#pragma unroll
      for (int e = 0; e < 8; ++e) vv[e] = (__bf16)t8[e];
      *(v8bf*)(Bs[buf] + k * SB_KN + n8) = vv;
    }
  };

  v8f acc00 = {}, acc01 = {}, acc10 = {}, acc11 = {};
  stage(0, 0);
  int buf = 0;
  for (int t = 0; t < total; ++t) {
    __syncthreads();
    if (t + 1 < total) stage(buf ^ 1, t + 1);
    v16bf a0 = frag_a_mk(As[buf], wm, lane);
    v16bf a1 = frag_a_mk(As[buf], wm + 16, lane);
    v16bf b0 = frag_b_kn(Bs[buf], wn, lane);
    v16bf b1 = frag_b_kn(Bs[buf], wn + 16, lane);
    acc00 = wmma_bf16(a0, b0, acc00);
    acc01 = wmma_bf16(a0, b1, acc01);
    acc10 = wmma_bf16(a1, b0, acc10);
    acc11 = wmma_bf16(a1, b1, acc11);
    buf ^= 1;
  }

  float*  oF = outF + (long long)bz * CI_ * NPIX;
  __bf16* oB = outB + (long long)bz * CI_ * NPIX;
  const int half = lane >> 4, col = lane & 15;
  const v8f* accs[4] = {&acc00, &acc01, &acc10, &acc11};
#pragma unroll
  for (int q = 0; q < 4; ++q) {
    int mBase = mBlock + wm + (q >> 1) * 16;
    int nOut  = nBlock + wn + (q & 1) * 16 + col;
    const v8f& a = *accs[q];
#pragma unroll
    for (int r = 0; r < 8; ++r) {
      int mOut = mBase + half * 8 + r;
      float sc = bn_g[mOut] * rsqrtf(bn_v[mOut] + 1e-5f);
      float v  = (a[r] + bias[mOut] - bn_m[mOut]) * sc + bn_b[mOut];
      v = fmaxf(v, 0.0f);
      long long idx = (long long)mOut * NPIX + nOut;
      oF[idx] = v;
      oB[idx] = (__bf16)v;
    }
  }
}

// ---------------------------------------------------------------------------
// Softmax over rows of 4096 (PAM) -> bf16
// ---------------------------------------------------------------------------
__global__ __launch_bounds__(256) void k_softmax_row4096(
    const float* __restrict__ E, __bf16* __restrict__ Out) {
  __shared__ float red[256];
  const long long row = blockIdx.x;
  const float* e = E + row * NPIX;
  __bf16* o = Out + row * NPIX;
  const int tid = threadIdx.x;
  float v[16];
  float mx = -3.4e38f;
#pragma unroll
  for (int i = 0; i < 16; ++i) { v[i] = e[tid + (i << 8)]; mx = fmaxf(mx, v[i]); }
  red[tid] = mx; __syncthreads();
  for (int s = 128; s > 0; s >>= 1) {
    if (tid < s) red[tid] = fmaxf(red[tid], red[tid + s]);
    __syncthreads();
  }
  mx = red[0]; __syncthreads();
  float sum = 0.0f;
#pragma unroll
  for (int i = 0; i < 16; ++i) { v[i] = __expf(v[i] - mx); sum += v[i]; }
  red[tid] = sum; __syncthreads();
  for (int s = 128; s > 0; s >>= 1) {
    if (tid < s) red[tid] += red[tid + s];
    __syncthreads();
  }
  const float inv = 1.0f / red[0];
#pragma unroll
  for (int i = 0; i < 16; ++i) o[tid + (i << 8)] = (__bf16)(v[i] * inv);
}

// ---------------------------------------------------------------------------
// CAM softmax: rows of 512; softmax(rowmax - v) == exp(rowmin - v)/sum
// ---------------------------------------------------------------------------
__global__ __launch_bounds__(256) void k_softmax_cam(
    const float* __restrict__ Cen, __bf16* __restrict__ Out) {
  __shared__ float red[256];
  const long long row = blockIdx.x;
  const float* e = Cen + row * CI_;
  __bf16* o = Out + row * CI_;
  const int tid = threadIdx.x;
  float v0 = e[tid], v1 = e[tid + 256];
  red[tid] = fminf(v0, v1); __syncthreads();
  for (int s = 128; s > 0; s >>= 1) {
    if (tid < s) red[tid] = fminf(red[tid], red[tid + s]);
    __syncthreads();
  }
  const float mn = red[0]; __syncthreads();
  float e0 = __expf(mn - v0), e1 = __expf(mn - v1);
  red[tid] = e0 + e1; __syncthreads();
  for (int s = 128; s > 0; s >>= 1) {
    if (tid < s) red[tid] += red[tid + s];
    __syncthreads();
  }
  const float inv = 1.0f / red[0];
  o[tid]       = (__bf16)(e0 * inv);
  o[tid + 256] = (__bf16)(e1 * inv);
}

// ---------------------------------------------------------------------------
// Elementwise / prep kernels
// ---------------------------------------------------------------------------
__global__ void k_gamma_add(const float* __restrict__ g, const float* __restrict__ a,
                            const float* __restrict__ f, float* __restrict__ o,
                            long long n) {
  long long i = (long long)blockIdx.x * 256 + threadIdx.x;
  if (i < n) o[i] = g[0] * a[i] + f[i];
}
__global__ void k_add_to_bf(const float* __restrict__ a, const float* __restrict__ b,
                            __bf16* __restrict__ o, long long n) {
  long long i = (long long)blockIdx.x * 256 + threadIdx.x;
  if (i < n) o[i] = (__bf16)(a[i] + b[i]);
}
__global__ void k_f2b(const float* __restrict__ a, __bf16* __restrict__ o, long long n) {
  long long i = (long long)blockIdx.x * 256 + threadIdx.x;
  if (i < n) o[i] = (__bf16)a[i];
}
// Pad classifier weights [rowsIn x cols] -> [64 x cols] (zeros beyond rowsIn)
__global__ void k_f2b_pad(const float* __restrict__ a, __bf16* __restrict__ o,
                          int rowsIn, int cols) {
  long long i = (long long)blockIdx.x * 256 + threadIdx.x;
  long long total = 64LL * cols;
  if (i >= total) return;
  int r = (int)(i / cols);
  o[i] = (r < rowsIn) ? (__bf16)a[i] : (__bf16)0.0f;
}
// OIHW fp32 3x3 weights -> bf16 [tap][oc][ic]
__global__ void k_w2b_tap(const float* __restrict__ a, __bf16* __restrict__ o,
                          int outC, int inC) {
  long long i = (long long)blockIdx.x * 256 + threadIdx.x;
  long long total = 9LL * outC * inC;
  if (i >= total) return;
  int ic = (int)(i % inC);
  long long r = i / inC;
  int oc = (int)(r % outC);
  int tap = (int)(r / outC);
  o[i] = (__bf16)a[((long long)oc * inC + ic) * 9 + tap];
}

// ---------------------------------------------------------------------------
// Launcher
// ---------------------------------------------------------------------------
template <bool TA, bool TB>
static void launch_gemm(const __bf16* A, const __bf16* B, float* Cf, __bf16* Cb,
                        const float* bias, int Mpad, int Mstore, int Ncols, int K,
                        int lda, int ldb, int ldc,
                        long long sA, long long sB, long long sC, hipStream_t s) {
  dim3 g(Ncols / 128, Mpad / 64, BATCH_);
  k_gemm_bf16<TA, TB><<<g, 256, 0, s>>>(A, B, Cf, Cb, bias, Mstore, K,
                                        lda, ldb, ldc, sA, sB, sC);
}

extern "C" void kernel_launch(void* const* d_in, const int* in_sizes, int n_in,
                              void* d_out, int out_size, void* d_ws, size_t ws_size,
                              hipStream_t stream) {
  (void)in_sizes; (void)n_in; (void)out_size; (void)ws_size;
  const float* x      = (const float*)d_in[0];
  const float* paW    = (const float*)d_in[1];
  const float* pab    = (const float*)d_in[2];
  const float* pa_g   = (const float*)d_in[3];
  const float* pa_b   = (const float*)d_in[4];
  const float* pa_m   = (const float*)d_in[5];
  const float* pa_v   = (const float*)d_in[6];
  const float* qW     = (const float*)d_in[7];
  const float* qb     = (const float*)d_in[8];
  const float* kW     = (const float*)d_in[9];
  const float* kb     = (const float*)d_in[10];
  const float* vW     = (const float*)d_in[11];
  const float* vb     = (const float*)d_in[12];
  const float* pgam   = (const float*)d_in[13];
  const float* paoW   = (const float*)d_in[14];
  const float* paob   = (const float*)d_in[15];
  const float* pao_g  = (const float*)d_in[16];
  const float* pao_b  = (const float*)d_in[17];
  const float* pao_m  = (const float*)d_in[18];
  const float* pao_v  = (const float*)d_in[19];
  const float* paclsW = (const float*)d_in[20];
  const float* paclsb = (const float*)d_in[21];
  const float* caW    = (const float*)d_in[22];
  const float* cab    = (const float*)d_in[23];
  const float* ca_g   = (const float*)d_in[24];
  const float* ca_b   = (const float*)d_in[25];
  const float* ca_m   = (const float*)d_in[26];
  const float* ca_v   = (const float*)d_in[27];
  const float* cgam   = (const float*)d_in[28];
  const float* caoW   = (const float*)d_in[29];
  const float* caob   = (const float*)d_in[30];
  const float* cao_g  = (const float*)d_in[31];
  const float* cao_b  = (const float*)d_in[32];
  const float* cao_m  = (const float*)d_in[33];
  const float* cao_v  = (const float*)d_in[34];
  const float* caclsW = (const float*)d_in[35];
  const float* caclsb = (const float*)d_in[36];
  const float* fW     = (const float*)d_in[37];
  const float* fb     = (const float*)d_in[38];

  // ---- workspace carving ----
  char* base = (char*)d_ws;
  size_t off = 0;
  auto carve = [&](size_t bytes) -> char* {
    char* p = base + off;
    off = (off + bytes + 255) & ~(size_t)255;
    return p;
  };
  const long long FEAT = (long long)BATCH_ * CI_ * NPIX;
  float*  f_pa    = (float*)carve(FEAT * 4);
  float*  f_ca    = (float*)carve(FEAT * 4);
  float*  pa_att  = (float*)carve(FEAT * 4);
  float*  pa_feat = (float*)carve(FEAT * 4);
  float*  pa_ft2  = (float*)carve(FEAT * 4);
  float*  ca_att  = (float*)carve(FEAT * 4);
  float*  ca_feat = (float*)carve(FEAT * 4);
  float*  ca_ft2  = (float*)carve(FEAT * 4);
  float*  energy  = (float*)carve((size_t)BATCH_ * NPIX * NPIX * 4);
  float*  cen     = (float*)carve((size_t)BATCH_ * CI_ * CI_ * 4);
  __bf16* f_pa_b  = (__bf16*)carve(FEAT * 2);
  __bf16* f_ca_b  = (__bf16*)carve(FEAT * 2);
  __bf16* q_b     = (__bf16*)carve((size_t)BATCH_ * C8_ * NPIX * 2);
  __bf16* k_b     = (__bf16*)carve((size_t)BATCH_ * C8_ * NPIX * 2);
  __bf16* v_b     = (__bf16*)carve(FEAT * 2);
  __bf16* attn_b  = (__bf16*)carve((size_t)BATCH_ * NPIX * NPIX * 2);
  __bf16* cattn_b = (__bf16*)carve((size_t)BATCH_ * CI_ * CI_ * 2);
  __bf16* pa2_b   = (__bf16*)carve(FEAT * 2);
  __bf16* ca2_b   = (__bf16*)carve(FEAT * 2);
  __bf16* fus_b   = (__bf16*)carve(FEAT * 2);
  __bf16* wq_b    = (__bf16*)carve((size_t)C8_ * CI_ * 2);
  __bf16* wk_b    = (__bf16*)carve((size_t)C8_ * CI_ * 2);
  __bf16* wv_b    = (__bf16*)carve((size_t)CI_ * CI_ * 2);
  __bf16* wpc_b   = (__bf16*)carve((size_t)64 * CI_ * 2);       // padded 19->64
  __bf16* wcc_b   = (__bf16*)carve((size_t)64 * CI_ * 2);
  __bf16* wf_b    = (__bf16*)carve((size_t)64 * CI_ * 2);
  __bf16* wpa_t   = (__bf16*)carve((size_t)9 * CI_ * CIN_ * 2); // [tap][oc][ic]
  __bf16* wca_t   = (__bf16*)carve((size_t)9 * CI_ * CIN_ * 2);
  __bf16* wpao_t  = (__bf16*)carve((size_t)9 * CI_ * CI_ * 2);
  __bf16* wcao_t  = (__bf16*)carve((size_t)9 * CI_ * CI_ * 2);

  float* out_fusion = (float*)d_out;
  float* out_pa = out_fusion + (long long)BATCH_ * NC_ * NPIX;
  float* out_ca = out_pa     + (long long)BATCH_ * NC_ * NPIX;

  auto conv = [&](const float* X, const __bf16* Wt, const float* bias,
                  const float* g, const float* bb, const float* mm,
                  const float* vv, float* oF, __bf16* oB, int inC, int nsSh) {
    dim3 gr(NPIX / 128, CI_ / 64, BATCH_);
    k_conv3x3_bnrelu<<<gr, 256, 0, stream>>>(X, Wt, bias, g, bb, mm, vv, oF, oB,
                                             inC, nsSh);
  };
  auto f2b = [&](const float* a, __bf16* o, long long n) {
    k_f2b<<<(unsigned)((n + 255) / 256), 256, 0, stream>>>(a, o, n);
  };
  auto f2b_pad = [&](const float* a, __bf16* o) {
    k_f2b_pad<<<(64 * CI_ + 255) / 256, 256, 0, stream>>>(a, o, NC_, CI_);
  };
  auto w2b = [&](const float* a, __bf16* o, int inC) {
    long long n = 9LL * CI_ * inC;
    k_w2b_tap<<<(unsigned)((n + 255) / 256), 256, 0, stream>>>(a, o, CI_, inC);
  };

  // ---- weight prep ----
  f2b(qW, wq_b, (long long)C8_ * CI_);
  f2b(kW, wk_b, (long long)C8_ * CI_);
  f2b(vW, wv_b, (long long)CI_ * CI_);
  f2b_pad(paclsW, wpc_b);
  f2b_pad(caclsW, wcc_b);
  f2b_pad(fW, wf_b);
  w2b(paW, wpa_t, CIN_);
  w2b(caW, wca_t, CIN_);
  w2b(paoW, wpao_t, CI_);
  w2b(caoW, wcao_t, CI_);

  const long long sF = (long long)CI_ * NPIX;
  const long long sQ = (long long)C8_ * NPIX;
  const long long sE = (long long)NPIX * NPIX;
  const long long sG = (long long)CI_ * CI_;
  const long long sO = (long long)NC_ * NPIX;

  // ---- PA branch ----
  conv(x, wpa_t, pab, pa_g, pa_b, pa_m, pa_v, f_pa, f_pa_b, CIN_, 6);
  launch_gemm<false, false>(wq_b, f_pa_b, nullptr, q_b, qb, C8_, C8_, NPIX, CI_,
                            CI_, NPIX, NPIX, 0, sF, sQ, stream);
  launch_gemm<false, false>(wk_b, f_pa_b, nullptr, k_b, kb, C8_, C8_, NPIX, CI_,
                            CI_, NPIX, NPIX, 0, sF, sQ, stream);
  launch_gemm<false, false>(wv_b, f_pa_b, nullptr, v_b, vb, CI_, CI_, NPIX, CI_,
                            CI_, NPIX, NPIX, 0, sF, sF, stream);
  // energy[n,m] = sum_c q[c,n]*k[c,m]   (A = q^T)
  launch_gemm<true, false>(q_b, k_b, energy, nullptr, nullptr, NPIX, NPIX, NPIX,
                           C8_, NPIX, NPIX, NPIX, sQ, sQ, sE, stream);
  k_softmax_row4096<<<BATCH_ * NPIX, 256, 0, stream>>>(energy, attn_b);
  // pa[c,i] = sum_j v[c,j]*attn[i,j]    (B = attn^T)
  launch_gemm<false, true>(v_b, attn_b, pa_att, nullptr, nullptr, CI_, CI_, NPIX,
                           NPIX, NPIX, NPIX, NPIX, sF, sE, sF, stream);
  k_gamma_add<<<(unsigned)((FEAT + 255) / 256), 256, 0, stream>>>(pgam, pa_att,
                                                                  f_pa, pa_feat, FEAT);
  conv(pa_feat, wpao_t, paob, pao_g, pao_b, pao_m, pao_v, pa_ft2, pa2_b, CI_, 4);
  launch_gemm<false, false>(wpc_b, pa2_b, out_pa, nullptr, paclsb, 64, NC_, NPIX,
                            CI_, CI_, NPIX, NPIX, 0, sF, sO, stream);

  // ---- CA branch ----
  conv(x, wca_t, cab, ca_g, ca_b, ca_m, ca_v, f_ca, f_ca_b, CIN_, 6);
  // cen[c,d] = sum_n gf[c,n]*gf[d,n]    (B = gf^T)
  launch_gemm<false, true>(f_ca_b, f_ca_b, cen, nullptr, nullptr, CI_, CI_, CI_,
                           NPIX, NPIX, NPIX, CI_, sF, sF, sG, stream);
  k_softmax_cam<<<BATCH_ * CI_, 256, 0, stream>>>(cen, cattn_b);
  // ca[c,n] = sum_d cattn[c,d]*gf[d,n]
  launch_gemm<false, false>(cattn_b, f_ca_b, ca_att, nullptr, nullptr, CI_, CI_,
                            NPIX, CI_, CI_, NPIX, NPIX, sG, sF, sF, stream);
  k_gamma_add<<<(unsigned)((FEAT + 255) / 256), 256, 0, stream>>>(cgam, ca_att,
                                                                  f_ca, ca_feat, FEAT);
  conv(ca_feat, wcao_t, caob, cao_g, cao_b, cao_m, cao_v, ca_ft2, ca2_b, CI_, 4);
  launch_gemm<false, false>(wcc_b, ca2_b, out_ca, nullptr, caclsb, 64, NC_, NPIX,
                            CI_, CI_, NPIX, NPIX, 0, sF, sO, stream);

  // ---- Fusion ----
  k_add_to_bf<<<(unsigned)((FEAT + 255) / 256), 256, 0, stream>>>(pa_ft2, ca_ft2,
                                                                  fus_b, FEAT);
  launch_gemm<false, false>(wf_b, fus_b, out_fusion, nullptr, fb, 64, NC_, NPIX,
                            CI_, CI_, NPIX, NPIX, 0, sF, sO, stream);
}
